// Block_66425964200172
// MI455X (gfx1250) — compile-verified, matrix operand order checked by reading, not run on hardware
//
#include <hip/hip_runtime.h>
#include <hip/hip_bf16.h>
#include <math.h>

// ---- types for WMMA fragments ----
typedef _Float16 v8h  __attribute__((ext_vector_type(8)));
typedef _Float16 v16h __attribute__((ext_vector_type(16)));
typedef float    v8f  __attribute__((ext_vector_type(8)));
typedef float    v4f  __attribute__((ext_vector_type(4)));

#define DIM      1024
#define HEADS    16
#define HEAD_DIM 64
#define HIDDEN   4096
#define SEQ      2048
#define BATCH    2
#define ROWS     (BATCH * SEQ)          // 4096
#define SCALE    0.125f                  // 64^-0.5
#define MASK_VAL -65504.0f
#define LN_EPS   1e-5f

// ---------------------------------------------------------------------------
// Fragment loaders. CDNA5 16-bit A (16x32) / B (32x16): lane L holds row/col
// (L%16); per lane the 16 halves are K = {h*8..h*8+7, 16+h*8..16+h*8+7},
// h = L/16  ->  two contiguous 16B (v8h) loads from row-major (rows,K).
// ---------------------------------------------------------------------------
__device__ __forceinline__ v16h frag_f16(const _Float16* __restrict__ p, int ld,
                                         int row, int k, int half) {
  const _Float16* base = p + (size_t)row * ld + (k + (half << 3));
  v8h lo = *(const v8h*)(base);
  v8h hi = *(const v8h*)(base + 16);
  return __builtin_shufflevector(lo, hi, 0,1,2,3,4,5,6,7,8,9,10,11,12,13,14,15);
}

// fp32 source (attention probabilities), non-temporal (537MB stream, read
// exactly once -> don't thrash the 192MB L2), convert to f16 inline.
__device__ __forceinline__ v16h frag_f32_nt(const float* __restrict__ p, int ld,
                                            int row, int k, int half) {
  const float* base = p + (size_t)row * ld + (k + (half << 3));
  v4f f0 = __builtin_nontemporal_load((const v4f*)(base));
  v4f f1 = __builtin_nontemporal_load((const v4f*)(base + 4));
  v4f f2 = __builtin_nontemporal_load((const v4f*)(base + 16));
  v4f f3 = __builtin_nontemporal_load((const v4f*)(base + 20));
  v16h r;
#pragma unroll
  for (int i = 0; i < 4; ++i) {
    r[i]      = (_Float16)f0[i];
    r[i + 4]  = (_Float16)f1[i];
    r[i + 8]  = (_Float16)f2[i];
    r[i + 12] = (_Float16)f3[i];
  }
  return r;
}

__device__ __forceinline__ v8f wmma(v16h a, v16h b, v8f c) {
  return __builtin_amdgcn_wmma_f32_16x16x32_f16(false, a, false, b, (short)0, c,
                                                false, false);
}

// ---------------------------------------------------------------------------
// Software-pipelined 32(M) x 64(N) wave-tile GEMM core, f16 x f16 -> f32.
// A pre-offset to the tile's first row (lda=K), B pre-offset to the tile's
// first output-column row (ldb=K). acc[j*2+rt] = (rt*16 rows, j*16 cols).
// Next iteration's 12 b128 loads issue before the current 8 WMMAs so the
// waitcnt pass overlaps load latency with matrix ops. Live set ~185 VGPRs;
// kernels using this core carry __launch_bounds__(256,2) to lift the default
// 128-VGPR cap (round-2/3 disasm showed folded spills at that cap).
// ---------------------------------------------------------------------------
__device__ __forceinline__ void core_f16_32x64(const _Float16* __restrict__ A,
                                               const _Float16* __restrict__ B,
                                               int K, int ml, int half,
                                               v8f acc[8]) {
  v16h a0 = frag_f16(A, K, ml, 0, half);
  v16h a1 = frag_f16(A, K, 16 + ml, 0, half);
  v16h b0 = frag_f16(B, K, ml, 0, half);
  v16h b1 = frag_f16(B, K, 16 + ml, 0, half);
  v16h b2 = frag_f16(B, K, 32 + ml, 0, half);
  v16h b3 = frag_f16(B, K, 48 + ml, 0, half);
  for (int kk = 32; kk < K; kk += 32) {
    v16h na0 = frag_f16(A, K, ml, kk, half);
    v16h na1 = frag_f16(A, K, 16 + ml, kk, half);
    v16h nb0 = frag_f16(B, K, ml, kk, half);
    v16h nb1 = frag_f16(B, K, 16 + ml, kk, half);
    v16h nb2 = frag_f16(B, K, 32 + ml, kk, half);
    v16h nb3 = frag_f16(B, K, 48 + ml, kk, half);
    acc[0] = wmma(a0, b0, acc[0]);
    acc[1] = wmma(a1, b0, acc[1]);
    acc[2] = wmma(a0, b1, acc[2]);
    acc[3] = wmma(a1, b1, acc[3]);
    acc[4] = wmma(a0, b2, acc[4]);
    acc[5] = wmma(a1, b2, acc[5]);
    acc[6] = wmma(a0, b3, acc[6]);
    acc[7] = wmma(a1, b3, acc[7]);
    a0 = na0; a1 = na1; b0 = nb0; b1 = nb1; b2 = nb2; b3 = nb3;
  }
  acc[0] = wmma(a0, b0, acc[0]);
  acc[1] = wmma(a1, b0, acc[1]);
  acc[2] = wmma(a0, b1, acc[2]);
  acc[3] = wmma(a1, b1, acc[3]);
  acc[4] = wmma(a0, b2, acc[4]);
  acc[5] = wmma(a1, b2, acc[5]);
  acc[6] = wmma(a0, b3, acc[6]);
  acc[7] = wmma(a1, b3, acc[7]);
}

// ---------------------------------------------------------------------------
// fp32 -> f16 conversion (weights)
// ---------------------------------------------------------------------------
__global__ void cvt_f16_kernel(const float* __restrict__ src,
                               _Float16* __restrict__ dst, int n) {
  int i = blockIdx.x * blockDim.x + threadIdx.x;
  if (i < n) dst[i] = (_Float16)src[i];
}

// ---------------------------------------------------------------------------
// LayerNorm over C=1024, one block (256 thr) per row, f16 output
// ---------------------------------------------------------------------------
__global__ void ln_kernel(const float* __restrict__ x,
                          const float* __restrict__ s,
                          const float* __restrict__ b,
                          _Float16* __restrict__ out) {
  __shared__ float r1[256];
  __shared__ float r2[256];
  int row = blockIdx.x;
  int t = threadIdx.x;
  const float* xr = x + (size_t)row * DIM;
  float v[4], a1 = 0.f, a2 = 0.f;
#pragma unroll
  for (int i = 0; i < 4; ++i) {
    v[i] = xr[t + i * 256];
    a1 += v[i];
    a2 += v[i] * v[i];
  }
  r1[t] = a1; r2[t] = a2;
  __syncthreads();
  for (int o = 128; o > 0; o >>= 1) {
    if (t < o) { r1[t] += r1[t + o]; r2[t] += r2[t + o]; }
    __syncthreads();
  }
  float mu  = r1[0] * (1.f / DIM);
  float var = r2[0] * (1.f / DIM) - mu * mu;
  float rstd = rsqrtf(var + LN_EPS);
  _Float16* orow = out + (size_t)row * DIM;
#pragma unroll
  for (int i = 0; i < 4; ++i) {
    int c = t + i * 256;
    orow[c] = (_Float16)((v[i] - mu) * rstd * s[c] + b[c]);
  }
}

// ---------------------------------------------------------------------------
// QKV GEMM: (4096,1024) x (3072,1024)^T, fused bias/scale, scatter into
// q(b,h,n,d) [pre-scaled], k(b,h,m,d), vT(b,h,d,m) as f16. 32x64 wave tiles.
// ---------------------------------------------------------------------------
__global__ void __launch_bounds__(256, 2)
gemm_qkv_kernel(const _Float16* __restrict__ A, const _Float16* __restrict__ W,
                const float* __restrict__ qb, const float* __restrict__ vb,
                _Float16* __restrict__ q, _Float16* __restrict__ k,
                _Float16* __restrict__ vT) {
  const int lane = threadIdx.x & 31, wid = threadIdx.x >> 5;
  const int ml = lane & 15, half = lane >> 4;
  int tile = blockIdx.x * 8 + wid;
  const int tiles_n = (3 * DIM) / 64;           // 48
  int tm = tile / tiles_n, tn = tile % tiles_n; // tm in [0,128)
  if (tm >= ROWS / 32) return;
  v8f acc[8] = {};
  core_f16_32x64(A + (size_t)tm * 32 * DIM, W + (size_t)tn * 64 * DIM, DIM,
                 ml, half, acc);
#pragma unroll
  for (int j = 0; j < 4; ++j) {
    int gn = tn * 64 + j * 16 + ml;
    int sel = gn >> 10, cc = gn & 1023;
    int hh = cc >> 6, dd = cc & 63;
#pragma unroll
    for (int rt = 0; rt < 2; ++rt) {
#pragma unroll
      for (int r = 0; r < 8; ++r) {
        int gm = tm * 32 + rt * 16 + r + half * 8;
        int bb = gm >> 11, n = gm & 2047;
        float val = acc[j * 2 + rt][r];
        size_t nd = (((size_t)(bb * HEADS + hh)) * SEQ + n) * HEAD_DIM + dd;
        if (sel == 0) {
          q[nd] = (_Float16)((val + qb[cc]) * SCALE);
        } else if (sel == 1) {
          k[nd] = (_Float16)val;
        } else {
          vT[(((size_t)(bb * HEADS + hh)) * HEAD_DIM + dd) * SEQ + n] =
              (_Float16)(val + vb[cc]);
        }
      }
    }
  }
}

// ---------------------------------------------------------------------------
// QK^T per (b,h): (2048,64) x (2048,64)^T, K=64, fused mask + additive bias,
// fp32 logits NT-stored straight into d_out's attn region. 32x64 wave tiles.
// ---------------------------------------------------------------------------
__global__ void __launch_bounds__(256, 2)
gemm_qk_kernel(const _Float16* __restrict__ q, const _Float16* __restrict__ kmat,
               const unsigned char* __restrict__ amask,
               const float* __restrict__ abias, float* __restrict__ logits) {
  const int lane = threadIdx.x & 31, wid = threadIdx.x >> 5;
  const int ml = lane & 15, half = lane >> 4;
  const int bh = blockIdx.y;
  const int bb = bh >> 4;
  const _Float16* Abase = q    + (size_t)bh * SEQ * HEAD_DIM;
  const _Float16* Bbase = kmat + (size_t)bh * SEQ * HEAD_DIM;
  float* outp = logits + (size_t)bh * SEQ * SEQ;
  int tile = blockIdx.x * 8 + wid;
  int tm = tile / 32, tn = tile % 32;           // 64 x 32 tiles
  v8f acc[8] = {};
  core_f16_32x64(Abase + (size_t)tm * 32 * HEAD_DIM,
                 Bbase + (size_t)tn * 64 * HEAD_DIM, HEAD_DIM, ml, half, acc);
#pragma unroll
  for (int j = 0; j < 4; ++j) {
    int gn = tn * 64 + j * 16 + ml;             // key index m
    bool masked = amask[bb * SEQ + gn] != 0;
#pragma unroll
    for (int rt = 0; rt < 2; ++rt) {
#pragma unroll
      for (int r = 0; r < 8; ++r) {
        int gm = tm * 32 + rt * 16 + r + half * 8;  // query index n
        float val = acc[j * 2 + rt][r];
        if (masked) val = MASK_VAL;
        val += abias[(size_t)gm * SEQ + gn];
        __builtin_nontemporal_store(val, &outp[(size_t)gm * SEQ + gn]);
      }
    }
  }
}

// ---------------------------------------------------------------------------
// Row softmax over 2048, in place in d_out attn region (fp32). The matrix is
// 537MB (~3x L2), touched once -> non-temporal both directions.
// ---------------------------------------------------------------------------
__global__ void softmax_kernel(float* __restrict__ attn) {
  __shared__ float red[256];
  float* p = attn + (size_t)blockIdx.x * SEQ;
  int t = threadIdx.x;
  float v[8], mx = -INFINITY;
#pragma unroll
  for (int i = 0; i < 8; ++i) {
    v[i] = __builtin_nontemporal_load(&p[t + i * 256]);
    mx = fmaxf(mx, v[i]);
  }
  red[t] = mx;
  __syncthreads();
  for (int o = 128; o > 0; o >>= 1) {
    if (t < o) red[t] = fmaxf(red[t], red[t + o]);
    __syncthreads();
  }
  float m = red[0];
  __syncthreads();
  float sm = 0.f;
#pragma unroll
  for (int i = 0; i < 8; ++i) {
    v[i] = expf(v[i] - m);
    sm += v[i];
  }
  red[t] = sm;
  __syncthreads();
  for (int o = 128; o > 0; o >>= 1) {
    if (t < o) red[t] += red[t + o];
    __syncthreads();
  }
  float inv = 1.f / red[0];
#pragma unroll
  for (int i = 0; i < 8; ++i)
    __builtin_nontemporal_store(v[i] * inv, &p[t + i * 256]);
}

// ---------------------------------------------------------------------------
// AV per (b,h): probs(fp32, 2048x2048) x vT(64,2048)^T -> out f16 (b,n,h,d).
// 16x64 wave tiles. Double-buffer only the streaming fp32 A operand (NT
// loads); the B operand (vT, 256 KB per bh) is L2/L0-hot, loaded just-in-time.
// ---------------------------------------------------------------------------
__global__ void __launch_bounds__(256, 2)
gemm_av_kernel(const float* __restrict__ probs, const _Float16* __restrict__ vT,
               _Float16* __restrict__ aout) {
  const int lane = threadIdx.x & 31, wid = threadIdx.x >> 5;
  const int ml = lane & 15, half = lane >> 4;
  const int bh = blockIdx.y;
  const int bb = bh >> 4, hh = bh & 15;
  const _Float16* Bb = vT + (size_t)bh * HEAD_DIM * SEQ;
  int tm = blockIdx.x * 8 + wid;                // 128 tiles, tn == 0
  const float* Ab = probs + (size_t)bh * SEQ * SEQ + (size_t)tm * 16 * SEQ;
  v8f acc[4] = {};
  v16h a = frag_f32_nt(Ab, SEQ, ml, 0, half);
  for (int kk = 32; kk < SEQ; kk += 32) {
    v16h na = frag_f32_nt(Ab, SEQ, ml, kk, half);
    acc[0] = wmma(a, frag_f16(Bb, SEQ, ml, kk - 32, half), acc[0]);
    acc[1] = wmma(a, frag_f16(Bb, SEQ, 16 + ml, kk - 32, half), acc[1]);
    acc[2] = wmma(a, frag_f16(Bb, SEQ, 32 + ml, kk - 32, half), acc[2]);
    acc[3] = wmma(a, frag_f16(Bb, SEQ, 48 + ml, kk - 32, half), acc[3]);
    a = na;
  }
  acc[0] = wmma(a, frag_f16(Bb, SEQ, ml, SEQ - 32, half), acc[0]);
  acc[1] = wmma(a, frag_f16(Bb, SEQ, 16 + ml, SEQ - 32, half), acc[1]);
  acc[2] = wmma(a, frag_f16(Bb, SEQ, 32 + ml, SEQ - 32, half), acc[2]);
  acc[3] = wmma(a, frag_f16(Bb, SEQ, 48 + ml, SEQ - 32, half), acc[3]);
#pragma unroll
  for (int j = 0; j < 4; ++j) {
    int dd = j * 16 + ml;
#pragma unroll
    for (int r = 0; r < 8; ++r) {
      int gm = tm * 16 + r + half * 8;          // query n
      aout[(((size_t)bb * SEQ + gm) * HEADS + hh) * HEAD_DIM + dd] =
          (_Float16)acc[j][r];
    }
  }
}

// ---------------------------------------------------------------------------
// Generic GEMM + bias + residual, fp32 output (proj and fc2). 32x64 tiles.
// ---------------------------------------------------------------------------
__global__ void __launch_bounds__(256, 2)
gemm_bias_res_kernel(const _Float16* __restrict__ A,
                     const _Float16* __restrict__ W,
                     const float* __restrict__ bias,
                     const float* __restrict__ res, float* __restrict__ out,
                     int M, int N, int K) {
  const int lane = threadIdx.x & 31, wid = threadIdx.x >> 5;
  const int ml = lane & 15, half = lane >> 4;
  int tile = blockIdx.x * 8 + wid;
  const int tiles_n = N >> 6;
  int tm = tile / tiles_n, tn = tile % tiles_n;
  if (tm >= (M >> 5)) return;
  v8f acc[8] = {};
  core_f16_32x64(A + (size_t)tm * 32 * K, W + (size_t)tn * 64 * K, K, ml, half,
                 acc);
#pragma unroll
  for (int j = 0; j < 4; ++j) {
    int gn = tn * 64 + j * 16 + ml;
    float bv = bias[gn];
#pragma unroll
    for (int rt = 0; rt < 2; ++rt) {
#pragma unroll
      for (int r = 0; r < 8; ++r) {
        int gm = tm * 32 + rt * 16 + r + half * 8;
        size_t idx = (size_t)gm * N + gn;
        out[idx] = acc[j * 2 + rt][r] + bv + res[idx];
      }
    }
  }
}

// ---------------------------------------------------------------------------
// FC1: GEMM + bias + exact GELU, f16 output (4096,4096,K=1024). 32x64 tiles.
// ---------------------------------------------------------------------------
__global__ void __launch_bounds__(256, 2)
gemm_gelu_kernel(const _Float16* __restrict__ A, const _Float16* __restrict__ W,
                 const float* __restrict__ bias, _Float16* __restrict__ out) {
  const int lane = threadIdx.x & 31, wid = threadIdx.x >> 5;
  const int ml = lane & 15, half = lane >> 4;
  int tile = blockIdx.x * 8 + wid;
  const int tiles_n = HIDDEN / 64;              // 64
  int tm = tile / tiles_n, tn = tile % tiles_n;
  if (tm >= ROWS / 32) return;
  v8f acc[8] = {};
  core_f16_32x64(A + (size_t)tm * 32 * DIM, W + (size_t)tn * 64 * DIM, DIM,
                 ml, half, acc);
#pragma unroll
  for (int j = 0; j < 4; ++j) {
    int gn = tn * 64 + j * 16 + ml;
    float bv = bias[gn];
#pragma unroll
    for (int rt = 0; rt < 2; ++rt) {
#pragma unroll
      for (int r = 0; r < 8; ++r) {
        int gm = tm * 32 + rt * 16 + r + half * 8;
        float v = acc[j * 2 + rt][r] + bv;
        float g = 0.5f * v * (1.f + erff(v * 0.70710678118f));
        out[(size_t)gm * HIDDEN + gn] = (_Float16)g;
      }
    }
  }
}

// ---------------------------------------------------------------------------
extern "C" void kernel_launch(void* const* d_in, const int* in_sizes, int n_in,
                              void* d_out, int out_size, void* d_ws,
                              size_t ws_size, hipStream_t stream) {
  const float* x      = (const float*)d_in[0];
  const unsigned char* amask = (const unsigned char*)d_in[1];  // jnp bool
  const float* abias  = (const float*)d_in[2];
  const float* qkv_w  = (const float*)d_in[3];
  const float* q_bias = (const float*)d_in[4];
  const float* v_bias = (const float*)d_in[5];
  const float* proj_w = (const float*)d_in[6];
  const float* proj_b = (const float*)d_in[7];
  const float* ln1_s  = (const float*)d_in[8];
  const float* ln1_b  = (const float*)d_in[9];
  const float* ln2_s  = (const float*)d_in[10];
  const float* ln2_b  = (const float*)d_in[11];
  const float* fc1_w  = (const float*)d_in[12];
  const float* fc1_b  = (const float*)d_in[13];
  const float* fc2_w  = (const float*)d_in[14];
  const float* fc2_b  = (const float*)d_in[15];

  float* out_x    = (float*)d_out;                         // (2,2048,1024)
  float* out_attn = out_x + (size_t)ROWS * DIM;            // (2,16,2048,2048)

  // ---- workspace carving (~120 MB) ----
  char* wsp = (char*)d_ws;
  auto carve16 = [&](size_t n) {
    _Float16* p = (_Float16*)wsp;
    wsp += n * sizeof(_Float16);
    return p;
  };
  _Float16* hf16  = carve16((size_t)ROWS * DIM);           // LN1 out
  _Float16* qf    = carve16((size_t)BATCH * HEADS * SEQ * HEAD_DIM);
  _Float16* kf    = carve16((size_t)BATCH * HEADS * SEQ * HEAD_DIM);
  _Float16* vT    = carve16((size_t)BATCH * HEADS * HEAD_DIM * SEQ);
  _Float16* aoutf = carve16((size_t)ROWS * DIM);           // attn output
  _Float16* h2f   = carve16((size_t)ROWS * DIM);           // LN2 out
  _Float16* hidf  = carve16((size_t)ROWS * HIDDEN);        // FC1 out
  _Float16* wq    = carve16((size_t)3 * DIM * DIM);
  _Float16* wp    = carve16((size_t)DIM * DIM);
  _Float16* w1    = carve16((size_t)HIDDEN * DIM);
  _Float16* w2    = carve16((size_t)DIM * HIDDEN);
  float* x1 = (float*)wsp;                                  // residual stream
  wsp += (size_t)ROWS * DIM * sizeof(float);

  // 1) weight conversion fp32 -> f16
  {
    int n;
    n = 3 * DIM * DIM;
    cvt_f16_kernel<<<(n + 255) / 256, 256, 0, stream>>>(qkv_w, wq, n);
    n = DIM * DIM;
    cvt_f16_kernel<<<(n + 255) / 256, 256, 0, stream>>>(proj_w, wp, n);
    n = HIDDEN * DIM;
    cvt_f16_kernel<<<(n + 255) / 256, 256, 0, stream>>>(fc1_w, w1, n);
    n = DIM * HIDDEN;
    cvt_f16_kernel<<<(n + 255) / 256, 256, 0, stream>>>(fc2_w, w2, n);
  }

  // 2) LN1
  ln_kernel<<<ROWS, 256, 0, stream>>>(x, ln1_s, ln1_b, hf16);

  // 3) QKV GEMM: 128 x 48 = 6144 tiles / 8 waves
  gemm_qkv_kernel<<<768, 256, 0, stream>>>(hf16, wq, q_bias, v_bias, qf, kf, vT);

  // 4) QK^T + mask + bias -> logits in d_out attn region (64x32 tiles per bh)
  gemm_qk_kernel<<<dim3(256, BATCH * HEADS), 256, 0, stream>>>(
      qf, kf, amask, abias, out_attn);

  // 5) softmax in place (B*H*N rows)
  softmax_kernel<<<BATCH * HEADS * SEQ, 256, 0, stream>>>(out_attn);

  // 6) AV -> attn output f16 (b,n,h,d): 128 tiles per bh
  gemm_av_kernel<<<dim3(16, BATCH * HEADS), 256, 0, stream>>>(out_attn, vT, aoutf);

  // 7) proj + bias + residual -> x1: 128 x 16 = 2048 tiles
  gemm_bias_res_kernel<<<256, 256, 0, stream>>>(aoutf, wp, proj_b, x, x1,
                                                ROWS, DIM, DIM);

  // 8) LN2
  ln_kernel<<<ROWS, 256, 0, stream>>>(x1, ln2_s, ln2_b, h2f);

  // 9) FC1 + GELU: 128 x 64 = 8192 tiles
  gemm_gelu_kernel<<<1024, 256, 0, stream>>>(h2f, w1, fc1_b, hidf);

  // 10) FC2 + bias + residual -> final x in d_out: 128 x 16 = 2048 tiles
  gemm_bias_res_kernel<<<256, 256, 0, stream>>>(hidf, w2, fc2_b, x1, out_x,
                                                ROWS, DIM, 4096);
}
